// GCNModel_3350074491436
// MI455X (gfx1250) — matrix operand order repaired and checked by reading, not compiled, not run on hardware
//
#include <hip/hip_runtime.h>

// ---------------------------------------------------------------------------
// GCN pipeline for gfx1250 (MI455X).
// bf16 activations + v_wmma_f32_16x16x32_bf16 GEMMs; fp32 atomic edge scatter.
// ---------------------------------------------------------------------------

typedef __attribute__((ext_vector_type(16))) __bf16 v16bf;
typedef __attribute__((ext_vector_type(8)))  __bf16 v8bf;
typedef __attribute__((ext_vector_type(8)))  float  v8f;

__device__ __forceinline__ unsigned short f2bf_bits(float f) {
  unsigned u = __builtin_bit_cast(unsigned, f);
  u += 0x7FFFu + ((u >> 16) & 1u);                 // round-to-nearest-even
  return (unsigned short)(u >> 16);
}
__device__ __forceinline__ __bf16 f2bf(float f) {
  unsigned short h = f2bf_bits(f);
  return __builtin_bit_cast(__bf16, h);
}
__device__ __forceinline__ float bf2f(unsigned short b) {
  return __builtin_bit_cast(float, (unsigned)b << 16);
}

// ---- A-fragment loaders (ISA 16-bit A layout: half0 K 0-7/16-23, half1 K 8-15/24-31)
__device__ __forceinline__ v16bf load_afrag(const float* p, float rs) {
  v16bf a;
#pragma unroll
  for (int i = 0; i < 8; ++i) a[i] = f2bf(p[i] * rs);
#pragma unroll
  for (int i = 0; i < 8; ++i) a[8 + i] = f2bf(p[16 + i] * rs);
  return a;
}
__device__ __forceinline__ v16bf load_afrag(const unsigned short* p, float) {
  v8bf lo = *(const v8bf*)p;          // 16B
  v8bf hi = *(const v8bf*)(p + 16);   // 16B
  v16bf a;
#pragma unroll
  for (int i = 0; i < 8; ++i) { a[i] = lo[i]; a[8 + i] = hi[i]; }
  return a;
}

__device__ __forceinline__ void storeC(float* C, size_t idx, float v) { C[idx] = v; }
__device__ __forceinline__ void storeC(unsigned short* C, size_t idx, float v) {
  C[idx] = f2bf_bits(v);
}

// ---------------- utility kernels ----------------

__global__ void k_zero(float* __restrict__ p, int n) {
  int i = blockIdx.x * blockDim.x + threadIdx.x;
  int stride = gridDim.x * blockDim.x;
  for (; i < n; i += stride) p[i] = 0.f;
}

__global__ void k_degree(const float* __restrict__ ew,
                         const int* __restrict__ src, const int* __restrict__ dst,
                         float* __restrict__ degO, float* __restrict__ degI,
                         float* __restrict__ cntS, float* __restrict__ cntD, int E) {
  int e = blockIdx.x * blockDim.x + threadIdx.x;
  if (e >= E) return;
  float w = ew[e];
  atomicAdd(&degO[src[e]], w);
  atomicAdd(&degI[dst[e]], w);
  atomicAdd(&cntS[src[e]], 1.f);
  atomicAdd(&cntD[dst[e]], 1.f);
}

__global__ void k_nodenorm(const float* __restrict__ cntS, const float* __restrict__ cntD,
                           float* __restrict__ cs, float* __restrict__ cd, int n) {
  int i = blockIdx.x * blockDim.x + threadIdx.x;
  if (i >= n) return;
  cs[i] = rsqrtf(fmaxf(cntS[i], 1.0f));
  cd[i] = rsqrtf(fmaxf(cntD[i], 1.0f));
}

__global__ void k_edgenorm(const float* __restrict__ ew,
                           const int* __restrict__ src, const int* __restrict__ dst,
                           const float* __restrict__ degO, const float* __restrict__ degI,
                           float* __restrict__ nEW, int E) {
  int e = blockIdx.x * blockDim.x + threadIdx.x;
  if (e >= E) return;
  float d = fmaxf(degO[src[e]], 1e-12f) * fmaxf(degI[dst[e]], 1e-12f);
  nEW[e] = ew[e] * rsqrtf(d);
}

// Pack row-major fp32 weight [K,Nc] into bf16 WMMA B-fragment layout.
__global__ void k_pack(const float* __restrict__ B, unsigned short* __restrict__ out,
                       int K, int Nc) {
  int total = K * Nc;
  int i = blockIdx.x * blockDim.x + threadIdx.x;
  if (i >= total) return;
  int j    = i & 15;
  int lane = (i >> 4) & 31;
  int rest = i >> 9;
  int kbn  = K >> 5;
  int kb   = rest % kbn;
  int nt   = rest / kbn;
  int k    = kb * 32 + ((lane >> 4) << 3) + (j & 7) + ((j >> 3) << 4);
  int col  = nt * 16 + (lane & 15);
  out[i] = f2bf_bits(B[(size_t)k * Nc + col]);
}

// ---------------- WMMA GEMM ----------------
// One wave computes a 16x32 C tile (two accumulators sharing the A fragment).
// grid = (M/16, Nc/32), block = 32.
// C = act( (A * rowscale[row]) @ B + bias ) * outscale[row]

template <typename AT, typename OT>
__global__ __launch_bounds__(32) void k_gemm(const AT* __restrict__ A,
                                             const unsigned short* __restrict__ Bp,
                                             const float* __restrict__ bias,
                                             const float* __restrict__ rowscale,
                                             const float* __restrict__ outscale,
                                             OT* __restrict__ C,
                                             int K, int Nc, int relu) {
  int lane = threadIdx.x;
  int half = lane >> 4, l = lane & 15;
  int mt = blockIdx.x, ntp = blockIdx.y;
  int row = mt * 16 + l;
  float rs = rowscale ? rowscale[row] : 1.0f;
  int col0 = ntp * 32 + l;
  float bv0 = bias ? bias[col0] : 0.0f;
  float bv1 = bias ? bias[col0 + 16] : 0.0f;
  v8f acc0, acc1;
#pragma unroll
  for (int i = 0; i < 8; ++i) { acc0[i] = bv0; acc1[i] = bv1; }

  int kbn = K >> 5;
  const unsigned short* bp0 = Bp + (size_t)(2 * ntp) * kbn * 512;
  const unsigned short* bp1 = bp0 + (size_t)kbn * 512;
  const AT* arow = A + (size_t)row * K;

  for (int kb = 0; kb < kbn; ++kb) {
    if (kb + 1 < kbn) __builtin_prefetch(arow + (kb + 1) * 32, 0, 3);
    v16bf a = load_afrag(arow + kb * 32 + 8 * half, rs);
    v16bf b0 = *(const v16bf*)(bp0 + ((size_t)(kb * 32 + lane) << 4));
    v16bf b1 = *(const v16bf*)(bp1 + ((size_t)(kb * 32 + lane) << 4));
    acc0 = __builtin_amdgcn_wmma_f32_16x16x32_bf16(false, a, false, b0,
                                                   (short)0, acc0, false, false);
    acc1 = __builtin_amdgcn_wmma_f32_16x16x32_bf16(false, a, false, b1,
                                                   (short)0, acc1, false, false);
  }

#pragma unroll
  for (int i = 0; i < 8; ++i) {
    int r = mt * 16 + i + 8 * half;
    float os = outscale ? outscale[r] : 1.0f;
    float v0 = acc0[i], v1 = acc1[i];
    if (relu) { v0 = fmaxf(v0, 0.f); v1 = fmaxf(v1, 0.f); }
    storeC(C, (size_t)r * Nc + col0, v0 * os);
    storeC(C, (size_t)r * Nc + col0 + 16, v1 * os);
  }
}

// ---------------- edge scatter: agg[dst] += w * h[src]  (h is bf16) ----------------

__global__ void k_scatter(const unsigned short* __restrict__ h,
                          const float* __restrict__ ew,
                          const int* __restrict__ src, const int* __restrict__ dst,
                          float* __restrict__ agg, int F) {
  int e = blockIdx.x;
  int t = threadIdx.x;
  int s = src[e], d = dst[e];
  float w = ew ? ew[e] : 1.0f;
  atomicAdd(&agg[(size_t)d * F + t], w * bf2f(h[(size_t)s * F + t]));
}

// h = bf16( relu(agg * c_dst[row] + bias[col]) * nextscale[row] )
__global__ void k_finish(const float* __restrict__ agg, const float* __restrict__ cd,
                         const float* __restrict__ bias, const float* __restrict__ nextscale,
                         unsigned short* __restrict__ out, int n, int F) {
  int i = blockIdx.x * blockDim.x + threadIdx.x;
  if (i >= n) return;
  int row = i / F, col = i % F;
  float v = fmaxf(agg[i] * cd[row] + bias[col], 0.f);
  if (nextscale) v *= nextscale[row];
  out[i] = f2bf_bits(v);
}

// ---------------- edge scorer: score[e] = [h[src]|h[dst]] @ cls_w + cls_b ----------------
// WMMA over E/16 row tiles, K=256 (src half then dst half), Nc=16; h is bf16 [N,128].

__global__ __launch_bounds__(32) void k_edge_score(const unsigned short* __restrict__ h,
                                                   const int* __restrict__ src,
                                                   const int* __restrict__ dst,
                                                   const unsigned short* __restrict__ Bp,
                                                   const float* __restrict__ bias,
                                                   float* __restrict__ out) {
  int lane = threadIdx.x;
  int half = lane >> 4, l = lane & 15;
  int mt = blockIdx.x;
  int row = mt * 16 + l;
  int s = src[row], d = dst[row];
  float bv = bias[l];
  v8f acc;
#pragma unroll
  for (int i = 0; i < 8; ++i) acc[i] = bv;

#pragma unroll
  for (int kb = 0; kb < 8; ++kb) {
    const unsigned short* base = (kb < 4) ? (h + (size_t)s * 128 + kb * 32)
                                          : (h + (size_t)d * 128 + (kb - 4) * 32);
    v16bf a = load_afrag(base + 8 * half, 1.0f);
    v16bf b = *(const v16bf*)(Bp + ((size_t)(kb * 32 + lane) << 4));
    acc = __builtin_amdgcn_wmma_f32_16x16x32_bf16(false, a, false, b,
                                                  (short)0, acc, false, false);
  }

#pragma unroll
  for (int i = 0; i < 8; ++i)
    out[(size_t)(mt * 16 + i + 8 * half) * 16 + l] = acc[i];
}

// ---------------------------------------------------------------------------

extern "C" void kernel_launch(void* const* d_in, const int* in_sizes, int n_in,
                              void* d_out, int out_size, void* d_ws, size_t ws_size,
                              hipStream_t stream) {
  (void)in_sizes; (void)n_in; (void)out_size; (void)ws_size;
  const int N = 10000, E = 160000;

  const float* X    = (const float*)d_in[0];
  const float* EW   = (const float*)d_in[1];
  const int*   SRC  = (const int*)d_in[2];
  const int*   DST  = (const int*)d_in[3];
  const float* LINW = (const float*)d_in[4];
  const float* LINB = (const float*)d_in[5];
  const float* W1   = (const float*)d_in[6];
  const float* B1   = (const float*)d_in[7];
  const float* W2   = (const float*)d_in[8];
  const float* B2   = (const float*)d_in[9];
  const float* W3   = (const float*)d_in[10];
  const float* B3   = (const float*)d_in[11];
  const float* W4   = (const float*)d_in[12];
  const float* B4   = (const float*)d_in[13];
  const float* CLSW = (const float*)d_in[14];
  const float* CLSB = (const float*)d_in[15];
  float* out = (float*)d_out;

  float* ws = (float*)d_ws;
  size_t o = 0;
  float* degO = ws + o; o += N;
  float* degI = ws + o; o += N;
  float* cntS = ws + o; o += N;
  float* cntD = ws + o; o += N;
  float* cs   = ws + o; o += N;
  float* cd   = ws + o; o += N;
  float* nEW  = ws + o; o += E;
  float* aggC = ws + o; o += (size_t)N * 256;                  // fp32 scatter accumulator
  unsigned short* hA = (unsigned short*)(ws + o); o += (size_t)N * 256;  // N*512 bf16
  unsigned short* hB = (unsigned short*)(ws + o); o += (size_t)N * 128;  // N*256 bf16
  unsigned short* pLIN = (unsigned short*)(ws + o);
  unsigned short* pW1  = pLIN + 128 * 256;
  unsigned short* pW2  = pW1  + 256 * 512;
  unsigned short* pW3  = pW2  + 512 * 256;
  unsigned short* pW4  = pW3  + 256 * 128;
  unsigned short* pCLS = pW4  + 128 * 128;

  // ---- norms ----
  k_zero<<<256, 256, 0, stream>>>(degO, 4 * N);
  k_degree<<<(E + 255) / 256, 256, 0, stream>>>(EW, SRC, DST, degO, degI, cntS, cntD, E);
  k_nodenorm<<<(N + 255) / 256, 256, 0, stream>>>(cntS, cntD, cs, cd, N);
  k_edgenorm<<<(E + 255) / 256, 256, 0, stream>>>(EW, SRC, DST, degO, degI, nEW, E);

  // ---- pack weights into WMMA B-fragment layout (bf16) ----
  k_pack<<<(128 * 256 + 255) / 256, 256, 0, stream>>>(LINW, pLIN, 128, 256);
  k_pack<<<(256 * 512 + 255) / 256, 256, 0, stream>>>(W1, pW1, 256, 512);
  k_pack<<<(512 * 256 + 255) / 256, 256, 0, stream>>>(W2, pW2, 512, 256);
  k_pack<<<(256 * 128 + 255) / 256, 256, 0, stream>>>(W3, pW3, 256, 128);
  k_pack<<<(128 * 128 + 255) / 256, 256, 0, stream>>>(W4, pW4, 128, 128);
  k_pack<<<(256 * 16 + 255) / 256, 256, 0, stream>>>(CLSW, pCLS, 256, 16);

  // ---- h0 = X @ lin_n_w + lin_n_b  (fp32 A -> bf16 out) ----
  k_gemm<float, unsigned short><<<dim3(N / 16, 256 / 32), 32, 0, stream>>>(
      X, pLIN, LINB, nullptr, nullptr, hB, 128, 256, 0);

  // ---- conv1: agg = seg(norm_ew * h0[src] -> dst); h1 = relu(agg @ w1 + b1) * cs ----
  k_zero<<<256, 256, 0, stream>>>(aggC, N * 256);
  k_scatter<<<E, 256, 0, stream>>>(hB, nEW, SRC, DST, aggC, 256);
  k_gemm<float, unsigned short><<<dim3(N / 16, 512 / 32), 32, 0, stream>>>(
      aggC, pW1, B1, nullptr, cs, hA, 256, 512, 1);

  // ---- conv2: t = h1 @ w2 (cs pre-folded); agg = seg(t[src]->dst); h2 = relu(agg*cd+b2)*cs ----
  k_gemm<unsigned short, unsigned short><<<dim3(N / 16, 256 / 32), 32, 0, stream>>>(
      hA, pW2, nullptr, nullptr, nullptr, hB, 512, 256, 0);
  k_zero<<<256, 256, 0, stream>>>(aggC, N * 256);
  k_scatter<<<E, 256, 0, stream>>>(hB, nullptr, SRC, DST, aggC, 256);
  k_finish<<<(N * 256 + 255) / 256, 256, 0, stream>>>(aggC, cd, B2, cs, hA, N * 256, 256);

  // ---- conv3 ----
  k_gemm<unsigned short, unsigned short><<<dim3(N / 16, 128 / 32), 32, 0, stream>>>(
      hA, pW3, nullptr, nullptr, nullptr, hB, 256, 128, 0);
  k_zero<<<256, 256, 0, stream>>>(aggC, N * 128);
  k_scatter<<<E, 128, 0, stream>>>(hB, nullptr, SRC, DST, aggC, 128);
  k_finish<<<(N * 128 + 255) / 256, 256, 0, stream>>>(aggC, cd, B3, cs, hA, N * 128, 128);

  // ---- conv4 (no next-layer scale: h4 feeds the scorer directly) ----
  k_gemm<unsigned short, unsigned short><<<dim3(N / 16, 128 / 32), 32, 0, stream>>>(
      hA, pW4, nullptr, nullptr, nullptr, hB, 128, 128, 0);
  k_zero<<<256, 256, 0, stream>>>(aggC, N * 128);
  k_scatter<<<E, 128, 0, stream>>>(hB, nullptr, SRC, DST, aggC, 128);
  k_finish<<<(N * 128 + 255) / 256, 256, 0, stream>>>(aggC, cd, B4, nullptr, hA, N * 128, 128);

  // ---- per-edge scores ----
  k_edge_score<<<E / 16, 32, 0, stream>>>(hA, SRC, DST, pCLS, CLSB, out);
}
